// TrajectorySplatAttention_43224550867611
// MI455X (gfx1250) — matrix operand order repaired and checked by reading, not compiled, not run on hardware
//
#include <hip/hip_runtime.h>

// TrajectorySplatAttention for MI455X (gfx1250, wave32).
// WMMA f16 16x16x32 for all matmuls; Tensor Data Mover (TDM) stages GEMM tiles.
// B=2, S=2048, D=1024, H=16, M=16, HD=64, WIN=8.

#define Bk 2
#define Sk 2048
#define Dk 1024
#define Hk 16
#define Mk 16
#define HDk 64
#define WINk 8

typedef _Float16 half_t;
typedef __attribute__((ext_vector_type(16))) _Float16 v16h;
typedef __attribute__((ext_vector_type(8)))  float    v8f;
typedef __attribute__((ext_vector_type(4)))  unsigned int v4u;
typedef __attribute__((ext_vector_type(8)))  int      v8i_t;
typedef __attribute__((ext_vector_type(4)))  int      v4i_t;

// ---------------------------------------------------------------------------
// TDM: 2D tile load Global -> LDS (descriptor per CDNA5 ISA
// §8.3/8.4). data_size=1 (2-byte elements). Optional LDS row padding.
// Issued per-wave (EXEC ignored); caller guards to one wave per block.
// ---------------------------------------------------------------------------
__device__ __forceinline__ void tdm_load_tile_f16(unsigned lds_off, const half_t* gptr,
                                                  unsigned tensor_d0, unsigned tensor_d1,
                                                  unsigned tile_d0, unsigned tile_d1,
                                                  unsigned stride0,
                                                  unsigned pad_interval, unsigned pad_amount_dw) {
    unsigned long long ga = (unsigned long long)(uintptr_t)gptr;
    v4u g0;
    g0[0] = 1u;                                          // count=1 (valid), no gather
    g0[1] = lds_off;                                     // lds_addr (bytes)
    g0[2] = (unsigned)(ga & 0xFFFFFFFFu);                // global_addr[31:0]
    g0[3] = (unsigned)((ga >> 32) & 0x01FFFFFFu)         // global_addr[56:32]
          | (2u << 30);                                  // type=2 ("image")
    v8i_t g1;
    unsigned w0 = (1u << 16);                            // data_size=1 -> 2 bytes
    if (pad_amount_dw) {
        w0 |= (1u << 20)                                 // pad_enable
            | (pad_interval << 22)                       // 2^(k+1) DWORDs per interval
            | ((pad_amount_dw - 1u) << 25);              // pad amount (DWORDs, minus 1)
    }
    g1[0] = (int)w0;
    g1[1] = (int)((tensor_d0 & 0xFFFFu) << 16);          // tensor_dim0[15:0]  -> bits 63:48
    g1[2] = (int)(((tensor_d0 >> 16) & 0xFFFFu)          // tensor_dim0[31:16] -> bits 79:64
                  | ((tensor_d1 & 0xFFFFu) << 16));      // tensor_dim1[15:0]  -> bits 95:80
    g1[3] = (int)(((tensor_d1 >> 16) & 0xFFFFu)          // tensor_dim1[31:16] -> bits 111:96
                  | (tile_d0 << 16));                    // tile_dim0          -> bits 127:112
    g1[4] = (int)tile_d1;                                // tile_dim1 (tile_dim2 = 0)
    g1[5] = (int)stride0;                                // tensor_dim0_stride[31:0]
    g1[6] = 0;                                           // stride0 hi / stride1 lo
    g1[7] = 0;                                           // stride1 hi
    v4i_t z4 = {};
#if defined(__clang_major__) && (__clang_major__ >= 23)
    v8i_t z8 = {};
    __builtin_amdgcn_tensor_load_to_lds(g0, g1, z4, z4, z8, 0);
#else
    __builtin_amdgcn_tensor_load_to_lds(g0, g1, z4, z4, 0);
#endif
}

// ---------------------------------------------------------------------------
// K1: normalized trajectories nt[b,j,:] = (x[j+1]-x[j]) / max(|.|,1e-8),
//     mw[b,j] = tanh(|.|).   One block per (b,j), 256 threads, 4 elems each.
// ---------------------------------------------------------------------------
__global__ __launch_bounds__(256) void traj_norm_kernel(const float* __restrict__ x,
                                                        float* __restrict__ nt,
                                                        float* __restrict__ mw) {
    int bj = blockIdx.x;
    int b = bj / (Sk - 1);
    int j = bj % (Sk - 1);
    const float* x0 = x + ((size_t)(b * Sk + j)) * Dk;
    const float* x1 = x0 + Dk;

    float tv[4];
    float ss = 0.f;
#pragma unroll
    for (int i = 0; i < 4; ++i) {
        int d = threadIdx.x + i * 256;
        tv[i] = x1[d] - x0[d];
        ss += tv[i] * tv[i];
    }
    __shared__ float red[256];
    red[threadIdx.x] = ss;
    __syncthreads();
    for (int s2 = 128; s2 > 0; s2 >>= 1) {
        if (threadIdx.x < s2) red[threadIdx.x] += red[threadIdx.x + s2];
        __syncthreads();
    }
    float mag = sqrtf(red[0]);
    float inv = 1.0f / fmaxf(mag, 1e-8f);
    float* dst = nt + ((size_t)(b * (Sk - 1) + j)) * Dk;
#pragma unroll
    for (int i = 0; i < 4; ++i) {
        int d = threadIdx.x + i * 256;
        dst[d] = tv[i] * inv;
    }
    if (threadIdx.x == 0) mw[b * (Sk - 1) + j] = tanhf(mag);
}

// ---------------------------------------------------------------------------
// K2: flows[b,p,:] = sum_{j in [p-WIN,p)} nw[p,j] * nt[b,j,:]
// ---------------------------------------------------------------------------
__global__ __launch_bounds__(256) void traj_flow_kernel(const float* __restrict__ nt,
                                                        const float* __restrict__ mw,
                                                        float* __restrict__ flows) {
    int bp = blockIdx.x;
    int b = bp / Sk;
    int p = bp % Sk;
    int jstart = (p >= WINk) ? (p - WINk) : 0;

    float wgt[WINk];
    float wsum = 0.f;
    int cnt = 0;
    for (int jj = jstart; jj < p; ++jj) {
        float w = mw[b * (Sk - 1) + jj] / (1.0f + (float)(p - 1 - jj));
        wgt[cnt++] = w;
        wsum += w;
    }
    float inv = 1.0f / fmaxf(wsum, 1e-8f);

    float* dst = flows + ((size_t)(b * Sk + p)) * Dk;
#pragma unroll
    for (int i = 0; i < 4; ++i) {
        int d = threadIdx.x + i * 256;
        float acc = 0.f;
        int c = 0;
        for (int jj = jstart; jj < p; ++jj)
            acc += wgt[c++] * nt[((size_t)(b * (Sk - 1) + jj)) * Dk + d];
        dst[d] = acc * inv;
    }
}

// ---------------------------------------------------------------------------
// f32 -> f16 conversion
// ---------------------------------------------------------------------------
__global__ __launch_bounds__(256) void f32_to_f16_kernel(const float* __restrict__ src,
                                                         half_t* __restrict__ dst, int n) {
    int t = blockIdx.x * 256 + threadIdx.x;
    if (t < n) dst[t] = (half_t)src[t];
}

// ---------------------------------------------------------------------------
// Tiled WMMA GEMM: C[MxN] (f32) = A[MxK] (f16) * B[KxN] (f16).
// Block tile 128x128, BK=32, 8 waves as 2x4, wave tile 64x32 (4x2 WMMA accs).
// Tiles staged by the Tensor Data Mover (one pair of tensor_load_to_lds per
// K-step, issued by wave 0, tracked via TENSORcnt).
// ---------------------------------------------------------------------------
__global__ __launch_bounds__(256) void gemm_f16_f32_kernel(const half_t* __restrict__ A,
                                                           const half_t* __restrict__ B,
                                                           float* __restrict__ C,
                                                           int M, int N, int K) {
    __shared__ __align__(16) half_t As[128][40];   // 32 + 8 pad halves (TDM-padded rows)
    __shared__ __align__(16) half_t Bs[32][136];   // 128 + 8 pad halves (TDM-padded rows)

    const int tid  = threadIdx.x;
    const int wave = tid >> 5;
    const int lane = tid & 31;
    const int lm = lane & 15;
    const int lh = lane >> 4;
    const int wm = wave >> 2;      // 0..1
    const int wn = wave & 3;       // 0..3
    const int row0 = blockIdx.x * 128;
    const int col0 = blockIdx.y * 128;

    const unsigned as_off = (unsigned)(uintptr_t)(&As[0][0]);
    const unsigned bs_off = (unsigned)(uintptr_t)(&Bs[0][0]);

    v8f acc[4][2] = {};

    for (int kk = 0; kk < K; kk += 32) {
        if (wave == 0) {
            // A tile: 128 rows x 32 halves; rows = 16 DWORDs + 4 DWORD pad.
            tdm_load_tile_f16(as_off, A + (size_t)row0 * K + kk,
                              (unsigned)K, (unsigned)M, 32u, 128u, (unsigned)K,
                              /*pad_interval 16dw=*/3u, /*pad 4dw=*/4u);
            // B tile: 32 rows x 128 halves; rows = 64 DWORDs + 4 DWORD pad.
            tdm_load_tile_f16(bs_off, B + (size_t)kk * N + col0,
                              (unsigned)N, (unsigned)K, 128u, 32u, (unsigned)N,
                              /*pad_interval 64dw=*/5u, /*pad 4dw=*/4u);
        }
        __builtin_amdgcn_s_wait_tensorcnt(0);
        __syncthreads();

        // Build A fragments (16x32 f16, A-layout: lane m=lm, K split by half).
        v16h af[4];
#pragma unroll
        for (int i = 0; i < 4; ++i) {
            int row = wm * 64 + i * 16 + lm;
#pragma unroll
            for (int r = 0; r < 8; ++r) {
                int kb = ((r < 4) ? 0 : 16) + 8 * lh + 2 * (r & 3);
                af[i][2 * r]     = As[row][kb];
                af[i][2 * r + 1] = As[row][kb + 1];
            }
        }
        // Build B fragments (32x16 f16, B-layout: lane n=lm, K split by half).
        v16h bf[2];
#pragma unroll
        for (int jb = 0; jb < 2; ++jb) {
            int col = wn * 32 + jb * 16 + lm;
#pragma unroll
            for (int r = 0; r < 8; ++r) {
                int k = ((r < 4) ? 0 : 16) + 8 * lh + 2 * (r & 3);
                bf[jb][2 * r]     = Bs[k][col];
                bf[jb][2 * r + 1] = Bs[k + 1][col];
            }
        }
#pragma unroll
        for (int i = 0; i < 4; ++i)
#pragma unroll
            for (int jb = 0; jb < 2; ++jb)
                acc[i][jb] = __builtin_amdgcn_wmma_f32_16x16x32_f16(
                    false, af[i], false, bf[jb], (short)0, acc[i][jb], false, false);
        __syncthreads();
    }

    // Store C (C-layout: n=lm, m=r+8*lh).
#pragma unroll
    for (int i = 0; i < 4; ++i)
#pragma unroll
        for (int jb = 0; jb < 2; ++jb)
#pragma unroll
            for (int r = 0; r < 8; ++r) {
                int m = 8 * lh + r;
                int row = row0 + wm * 64 + i * 16 + m;
                int col = col0 + wn * 32 + jb * 16 + lm;
                C[(size_t)row * N + col] = acc[i][jb][r];
            }
}

// ---------------------------------------------------------------------------
// Splat weights: qw/kw[b,h,s,m] = sigmoid(amp) * exp(-0.5/(scale^2+1e-8) *
//                ||(q_or_k + sigmoid(gate)*traj) - center||^2), stored f16.
// ---------------------------------------------------------------------------
__global__ __launch_bounds__(256) void splat_kernel(const float* __restrict__ qkv,
                                                    const float* __restrict__ flows,
                                                    const float* __restrict__ centers,
                                                    const float* __restrict__ log_scales,
                                                    const float* __restrict__ amplitudes,
                                                    const float* __restrict__ gate_p,
                                                    half_t* __restrict__ qwh,
                                                    half_t* __restrict__ kwh) {
    int t = blockIdx.x * 256 + threadIdx.x;
    int m  = t & 15;   t >>= 4;
    int s  = t & 2047; t >>= 11;
    int h  = t & 15;   t >>= 4;
    int qk = t & 1;    t >>= 1;
    int b  = t;

    float gate  = 1.0f / (1.0f + __expf(-gate_p[0]));
    float scale = __expf(log_scales[h * Mk + m]);
    float invs  = -0.5f / (scale * scale + 1e-8f);
    float amp   = 1.0f / (1.0f + __expf(-amplitudes[h * Mk + m]));

    const float* tq = qkv + ((size_t)(b * Sk + s)) * (3 * Dk) + qk * Dk + h * HDk;
    const float* tf = flows + ((size_t)(b * Sk + s)) * Dk + h * HDk;
    const float* cc = centers + (size_t)(h * Mk + m) * HDk;

    float d2 = 0.f;
#pragma unroll
    for (int d = 0; d < HDk; ++d) {
        float e = tq[d] + gate * tf[d] - cc[d];
        d2 += e * e;
    }
    float w = amp * __expf(invs * d2);
    half_t* dst = qk ? kwh : qwh;
    dst[(((size_t)(b * Hk + h)) * Sk + s) * Mk + m] = (half_t)w;
}

// ---------------------------------------------------------------------------
// V transpose+convert: vht[b,h,d,s] (f16) from qkv v-part.
// ---------------------------------------------------------------------------
__global__ __launch_bounds__(256) void vconv_kernel(const float* __restrict__ qkv,
                                                    half_t* __restrict__ vht) {
    int t = blockIdx.x * 256 + threadIdx.x;
    int c = t & 1023;
    int s = (t >> 10) & 2047;
    int b = t >> 21;
    int h = c >> 6, d = c & 63;
    vht[(((size_t)(b * Hk + h)) * HDk + d) * Sk + s] =
        (half_t)qkv[((size_t)(b * Sk + s)) * (3 * Dk) + 2 * Dk + c];
}

// ---------------------------------------------------------------------------
// Flash attention over splat scores. One wave per 16-row i-tile.
// scores = qw(16x16, K padded to 32) @ kw^T; p = exp(s) (no max-sub needed:
// 0 <= s <= 16*sigmoid(1)^2 ~ 8.6); O += p @ V via 4 WMMAs; normalize by
// row-sums reduced with shfl_xor. P transposed C->A layout through LDS.
// ---------------------------------------------------------------------------
__global__ __launch_bounds__(256) void flash_kernel(const half_t* __restrict__ qwh,
                                                    const half_t* __restrict__ kwh,
                                                    const half_t* __restrict__ vht,
                                                    half_t* __restrict__ attn_h) {
    __shared__ __align__(16) float pT[8][16 * 32];   // per-wave 2KB transpose buffer

    const int tid  = threadIdx.x;
    const int wave = tid >> 5;
    const int lane = tid & 31;
    const int lm = lane & 15;
    const int lh = lane >> 4;

    const int bh = blockIdx.y;            // b*H + h
    const int b  = bh / Hk;
    const int h  = bh % Hk;
    const int i0 = blockIdx.x * 128 + wave * 16;

    const size_t qbase = ((size_t)bh) * Sk * Mk;
    const size_t vbase = ((size_t)bh) * HDk * Sk;
    float* myP = &pT[wave][0];

    // Q fragment: A-layout 16x32, real K = 0..15, upper 16 zero-padded.
    v16h qa;
    {
        const half_t* qr = &qwh[qbase + (size_t)(i0 + lm) * Mk];
#pragma unroll
        for (int i = 0; i < 16; ++i) qa[i] = (lh == 0) ? qr[i] : (half_t)0.f;
    }

    v8f o[4] = {};
    float rs[8] = {0.f, 0.f, 0.f, 0.f, 0.f, 0.f, 0.f, 0.f};

    for (int j0 = 0; j0 < Sk; j0 += 32) {
        // K fragments (B-layout 32x16): k = m-index, half==1 rows are K-pad.
        v16h kb0, kb1;
        if (lh == 0) {
            const half_t* k0 = &kwh[qbase + (size_t)(j0 + lm) * Mk];
            const half_t* k1 = &kwh[qbase + (size_t)(j0 + 16 + lm) * Mk];
#pragma unroll
            for (int i = 0; i < 16; ++i) { kb0[i] = k0[i]; kb1[i] = k1[i]; }
        } else {
#pragma unroll
            for (int i = 0; i < 16; ++i) { kb0[i] = (half_t)0.f; kb1[i] = (half_t)0.f; }
        }

        v8f z = {};
        v8f s0 = __builtin_amdgcn_wmma_f32_16x16x32_f16(false, qa, false, kb0, (short)0, z, false, false);
        v8f s1 = __builtin_amdgcn_wmma_f32_16x16x32_f16(false, qa, false, kb1, (short)0, z, false, false);

        // exp + row-sum partials + write P to LDS (C-layout -> row-major tile).
#pragma unroll
        for (int r = 0; r < 8; ++r) {
            float p0 = __expf(s0[r]);
            float p1 = __expf(s1[r]);
            rs[r] += p0 + p1;
            int row = r + 8 * lh;
            myP[row * 32 + lm]      = p0;
            myP[row * 32 + 16 + lm] = p1;
        }

        // Re-gather P in A-layout (16x32 f16).
        v16h pa;
#pragma unroll
        for (int r = 0; r < 8; ++r) {
            int kb = 16 * lh + 2 * r;
            pa[2 * r]     = (half_t)myP[lm * 32 + kb];
            pa[2 * r + 1] = (half_t)myP[lm * 32 + kb + 1];
        }

        // V fragments (B-layout 32x16) from transposed V: 16 contiguous halves.
#pragma unroll
        for (int t4 = 0; t4 < 4; ++t4) {
            const half_t* pv = &vht[vbase + (size_t)(t4 * 16 + lm) * Sk + j0 + 16 * lh];
            v16h vb;
#pragma unroll
            for (int i = 0; i < 16; ++i) vb[i] = pv[i];
            o[t4] = __builtin_amdgcn_wmma_f32_16x16x32_f16(false, pa, false, vb, (short)0, o[t4], false, false);
        }
    }

    // Reduce row-sums across the 16 lanes of each half-group.
#pragma unroll
    for (int r = 0; r < 8; ++r) {
#pragma unroll
        for (int off = 1; off < 16; off <<= 1) rs[r] += __shfl_xor(rs[r], off, 32);
        rs[r] = 1.0f / rs[r];
    }

    // Normalize and store attn output as f16 [B*S, D] rows for the out-proj GEMM.
#pragma unroll
    for (int t4 = 0; t4 < 4; ++t4)
#pragma unroll
        for (int r = 0; r < 8; ++r) {
            int row = i0 + 8 * lh + r;
            int col = h * HDk + t4 * 16 + lm;
            attn_h[((size_t)(b * Sk + row)) * Dk + col] = (half_t)(o[t4][r] * rs[r]);
        }
}

// ---------------------------------------------------------------------------
// Host launcher
// ---------------------------------------------------------------------------
extern "C" void kernel_launch(void* const* d_in, const int* in_sizes, int n_in,
                              void* d_out, int out_size, void* d_ws, size_t ws_size,
                              hipStream_t stream) {
    const float* x          = (const float*)d_in[0];
    const float* w_qkv      = (const float*)d_in[1];
    const float* w_out      = (const float*)d_in[2];
    const float* centers    = (const float*)d_in[3];
    const float* log_scales = (const float*)d_in[4];
    const float* amplitudes = (const float*)d_in[5];
    const float* gate       = (const float*)d_in[6];
    float* out = (float*)d_out;

    char* ws = (char*)d_ws;
    size_t off = 0;
    auto bump = [&](size_t bytes) -> size_t {
        size_t o = off;
        off = (off + bytes + 255) & ~(size_t)255;
        return o;
    };

    const size_t BS = (size_t)Bk * Sk;                    // 4096
    float*  nt     = (float*) (ws + bump((size_t)Bk * (Sk - 1) * Dk * 4));
    float*  mw     = (float*) (ws + bump((size_t)Bk * (Sk - 1) * 4));
    float*  flows  = (float*) (ws + bump(BS * Dk * 4));
    half_t* x_h    = (half_t*)(ws + bump(BS * Dk * 2));
    half_t* wqkv_h = (half_t*)(ws + bump((size_t)Dk * 3 * Dk * 2));
    half_t* wout_h = (half_t*)(ws + bump((size_t)Dk * Dk * 2));
    float*  qkv    = (float*) (ws + bump(BS * 3 * Dk * 4));
    half_t* qwh    = (half_t*)(ws + bump((size_t)Bk * Hk * Sk * Mk * 2));
    half_t* kwh    = (half_t*)(ws + bump((size_t)Bk * Hk * Sk * Mk * 2));
    half_t* vht    = (half_t*)(ws + bump((size_t)Bk * Hk * HDk * Sk * 2));
    half_t* attn_h = (half_t*)(ws + bump(BS * Dk * 2));
    (void)ws_size; (void)in_sizes; (void)n_in; (void)out_size;

    // 1) trajectory flows
    traj_norm_kernel<<<Bk * (Sk - 1), 256, 0, stream>>>(x, nt, mw);
    traj_flow_kernel<<<Bk * Sk, 256, 0, stream>>>(nt, mw, flows);

    // 2) f16 conversions
    {
        int n1 = (int)(BS * Dk);
        f32_to_f16_kernel<<<(n1 + 255) / 256, 256, 0, stream>>>(x, x_h, n1);
        int n2 = Dk * 3 * Dk;
        f32_to_f16_kernel<<<(n2 + 255) / 256, 256, 0, stream>>>(w_qkv, wqkv_h, n2);
        int n3 = Dk * Dk;
        f32_to_f16_kernel<<<(n3 + 255) / 256, 256, 0, stream>>>(w_out, wout_h, n3);
    }

    // 3) qkv = x @ w_qkv   (4096 x 3072 x 1024)
    gemm_f16_f32_kernel<<<dim3((int)BS / 128, (3 * Dk) / 128), 256, 0, stream>>>(
        x_h, wqkv_h, qkv, (int)BS, 3 * Dk, Dk);

    // 4) splat weights + V transpose
    {
        int nth = 2 * Bk * Hk * Sk * Mk;
        splat_kernel<<<nth / 256, 256, 0, stream>>>(qkv, flows, centers, log_scales,
                                                    amplitudes, gate, qwh, kwh);
        int nv = (int)(BS * Dk);
        vconv_kernel<<<nv / 256, 256, 0, stream>>>(qkv, vht);
    }

    // 5) flash splat-attention
    flash_kernel<<<dim3(Sk / 128, Bk * Hk), 256, 0, stream>>>(qwh, kwh, vht, attn_h);

    // 6) out = attn @ w_out   (4096 x 1024 x 1024)
    gemm_f16_f32_kernel<<<dim3((int)BS / 128, Dk / 128), 256, 0, stream>>>(
        attn_h, wout_h, out, (int)BS, Dk, Dk);
}
